// SGCNet_41308995452969
// MI455X (gfx1250) — compile-verified, matrix operand order checked by reading, not run on hardware
//
#include <hip/hip_runtime.h>
#include <hip/hip_bf16.h>

typedef __bf16 bf16;
typedef __attribute__((ext_vector_type(16))) __bf16 v16bf;
typedef __attribute__((ext_vector_type(8)))  float  v8f;

#define N_NODES  100000
#define N_EDGES  1600000
#define IN_DIM   256
#define H_DIM    128
#define N_TILES  (N_NODES / 16)   // 6250 exact
#define BN_EPS   1e-5f

// ---------------------------------------------------------------------------
// WMMA wrapper: D = A(16x32 bf16) * B(32x16 bf16) + C(16x16 f32)
// ---------------------------------------------------------------------------
__device__ __forceinline__ v8f wmma_bf16(v16bf a, v16bf b, v8f c) {
  return __builtin_amdgcn_wmma_f32_16x16x32_bf16(false, a, false, b, (short)0, c,
                                                 false, false);
}

// A fragment (16x32) from row-major fp32 [row stride S], optional per-row scale.
// ISA 16-bit A layout: lanes 0-15 hold row m with K = kb+{0..7, 16..23};
// lanes 16-31 hold row m with K = kb+{8..15, 24..31}.
__device__ __forceinline__ v16bf a_frag_f32(const float* __restrict__ src, int S,
                                            int rowBase, int kb,
                                            const float* __restrict__ rowScale,
                                            int lane) {
  int m = lane & 15, half = lane >> 4;
  const float* p = src + (rowBase + m) * S + kb + half * 8;
  float s = rowScale ? rowScale[rowBase + m] : 1.0f;
  v16bf a;
#pragma unroll
  for (int i = 0; i < 8; ++i) a[i] = (bf16)(s * p[i]);
#pragma unroll
  for (int i = 0; i < 8; ++i) a[8 + i] = (bf16)(s * p[16 + i]);
  return a;
}

// A fragment (16x32) from bf16 LDS tile [row stride S].
__device__ __forceinline__ v16bf a_frag_bf16(const bf16* __restrict__ src, int S,
                                             int kb, int lane) {
  int m = lane & 15, half = lane >> 4;
  const bf16* p = src + m * S + kb + half * 8;
  v16bf a;
#pragma unroll
  for (int i = 0; i < 8; ++i) a[i] = p[i];
#pragma unroll
  for (int i = 0; i < 8; ++i) a[8 + i] = p[16 + i];
  return a;
}

// B fragment (32x16) from *transposed* bf16 weights Wt[Ncols][K]:
// lane holds column nb+m; contiguous 8-element K runs per half.
__device__ __forceinline__ v16bf b_frag(const bf16* __restrict__ Wt, int K, int nb,
                                        int kb, int lane) {
  int m = lane & 15, half = lane >> 4;
  const bf16* p = Wt + (nb + m) * K + kb + half * 8;
  v16bf b;
#pragma unroll
  for (int i = 0; i < 8; ++i) b[i] = p[i];
#pragma unroll
  for (int i = 0; i < 8; ++i) b[8 + i] = p[16 + i];
  return b;
}

// ---------------------------------------------------------------------------
// 0) Convert + transpose all weights to bf16 (tiny, once per call)
// ---------------------------------------------------------------------------
__global__ void prep_weights(const float* __restrict__ We, const float* __restrict__ Wc,
                             const float* __restrict__ W1, const float* __restrict__ W2,
                             const float* __restrict__ W3,
                             bf16* __restrict__ WeT, bf16* __restrict__ WcT,
                             bf16* __restrict__ W1T, bf16* __restrict__ W2T,
                             bf16* __restrict__ W3T) {
  int idx = blockIdx.x * 256 + threadIdx.x;
  if (idx < 32768) { int k = idx >> 7, n = idx & 127; WeT[n * 256 + k] = (bf16)We[k * 128 + n]; return; }
  idx -= 32768;
  if (idx < 16384) { int k = idx >> 7, n = idx & 127; WcT[n * 128 + k] = (bf16)Wc[k * 128 + n]; return; }
  idx -= 16384;
  if (idx < 8192)  { int k = idx >> 6, n = idx & 63;  W1T[n * 128 + k] = (bf16)W1[k * 64 + n]; return; }
  idx -= 8192;
  if (idx < 2048)  { int k = idx >> 5, n = idx & 31;  W2T[n * 64 + k]  = (bf16)W2[k * 32 + n]; return; }
  idx -= 2048;
  if (idx < 1024)  { int n = idx >> 5, k = idx & 31;  // W3 padded 32x18 -> 32x32
    W3T[n * 32 + k] = (n < 18) ? (bf16)W3[k * 18 + n] : (bf16)0.0f; }
}

// ---------------------------------------------------------------------------
// 1) Embedding: h = BN(relu(x @ We + be)) -> bf16 [N,128]
//    One wave per 16-row tile; 8 col-tiles x 8 k-steps of bf16 WMMA.
// ---------------------------------------------------------------------------
__global__ void embed_kernel(const float* __restrict__ x, const bf16* __restrict__ WeT,
                             const float* __restrict__ be, const float* __restrict__ gamma,
                             const float* __restrict__ beta, const float* __restrict__ mean,
                             const float* __restrict__ var, bf16* __restrict__ h) {
  int tile = blockIdx.x * 8 + (threadIdx.x >> 5);
  if (tile >= N_TILES) return;
  int lane = threadIdx.x & 31, m = lane & 15, half = lane >> 4;
  int rowBase = tile * 16;

  v16bf A[8];
#pragma unroll
  for (int kk = 0; kk < 8; ++kk)
    A[kk] = a_frag_f32(x, IN_DIM, rowBase, kk * 32, nullptr, lane);

#pragma unroll
  for (int ct = 0; ct < 8; ++ct) {
    v8f acc = {};
#pragma unroll
    for (int kk = 0; kk < 8; ++kk)
      acc = wmma_bf16(A[kk], b_frag(WeT, IN_DIM, ct * 16, kk * 32, lane), acc);
    int col = ct * 16 + m;
    float bias = be[col];
    float s  = gamma[col] * rsqrtf(var[col] + BN_EPS);
    float sh = beta[col] - mean[col] * s;
#pragma unroll
    for (int r = 0; r < 8; ++r) {          // C layout: VGPR r -> row r (+8 for hi lanes)
      float t = acc[r] + bias;
      t = fmaxf(t, 0.0f);
      t = t * s + sh;
      h[(rowBase + r + half * 8) * H_DIM + col] = (bf16)t;
    }
  }
}

// ---------------------------------------------------------------------------
// 2) Degree / dinv / self-loop init
// ---------------------------------------------------------------------------
__global__ void init_deg(unsigned* __restrict__ deg) {
  int i = blockIdx.x * 256 + threadIdx.x;
  if (i < N_NODES) deg[i] = 1u;            // self-loop
}

__global__ void degree_kernel(const int* __restrict__ ei, unsigned* __restrict__ deg) {
  int e = blockIdx.x * 256 + threadIdx.x;
  if (e < N_EDGES) atomicAdd(&deg[ei[N_EDGES + e]], 1u);   // col indices
}

__global__ void dinv_kernel(const unsigned* __restrict__ deg, float* __restrict__ dinv) {
  int i = blockIdx.x * 256 + threadIdx.x;
  if (i < N_NODES) dinv[i] = rsqrtf((float)deg[i]);
}

// agg[c,:] = dinv[c] * h[c,:]   (self-loop term; also resets agg every call)
__global__ void agg_init(const bf16* __restrict__ h, const float* __restrict__ dinv,
                         float* __restrict__ agg) {
  int idx = blockIdx.x * 256 + threadIdx.x;
  if (idx < N_NODES * H_DIM) agg[idx] = dinv[idx >> 7] * (float)h[idx];
}

// ---------------------------------------------------------------------------
// 3) Edge scatter: agg[c,:] += dinv[r] * h[r,:]  (one wave per edge, 4 f32/lane)
// ---------------------------------------------------------------------------
__global__ void scatter_kernel(const int* __restrict__ ei, const bf16* __restrict__ h,
                               const float* __restrict__ dinv, float* __restrict__ agg) {
  int tid = blockIdx.x * 256 + threadIdx.x;
  int e = tid >> 5;
  if (e >= N_EDGES) return;
  int lane = tid & 31;
  int r = ei[e], c = ei[N_EDGES + e];
  float w = dinv[r];
  int j = lane * 4;
  const bf16* hp = h + r * H_DIM + j;
  float* ap = agg + c * H_DIM + j;
#pragma unroll
  for (int k = 0; k < 4; ++k)
    __hip_atomic_fetch_add(ap + k, w * (float)hp[k],
                           __ATOMIC_RELAXED, __HIP_MEMORY_SCOPE_AGENT);
}

// ---------------------------------------------------------------------------
// 4) Fused head: scale by dinv[c], GEMM chain 128->128->64->32->18, sigmoid.
//    One wave per 16-row tile; inter-stage activations staged in per-wave LDS.
// ---------------------------------------------------------------------------
__global__ void head_kernel(const float* __restrict__ agg, const float* __restrict__ dinv,
                            const bf16* __restrict__ WcT, const float* __restrict__ bc,
                            const bf16* __restrict__ W1T, const float* __restrict__ b1,
                            const bf16* __restrict__ W2T, const float* __restrict__ b2,
                            const bf16* __restrict__ W3T, const float* __restrict__ b3,
                            float* __restrict__ out) {
  __shared__ bf16 lt[8][16 * H_DIM];   // 32 KB
  __shared__ bf16 lu[8][16 * 64];      // 16 KB
  __shared__ bf16 lv[8][16 * 32];      //  8 KB
  int w = threadIdx.x >> 5;
  int tile = blockIdx.x * 8 + w;
  if (tile >= N_TILES) return;         // no block barriers -> safe early exit
  int lane = threadIdx.x & 31, m = lane & 15, half = lane >> 4;
  int rowBase = tile * 16;

  // Stage A: agg rows pre-scaled by dinv (completes the symmetric norm)
  v16bf A[4];
#pragma unroll
  for (int kk = 0; kk < 4; ++kk)
    A[kk] = a_frag_f32(agg, H_DIM, rowBase, kk * 32, dinv, lane);

  // GEMM1: [16x128] x Wc[128x128] -> relu -> lt
#pragma unroll
  for (int ct = 0; ct < 8; ++ct) {
    v8f acc = {};
#pragma unroll
    for (int kk = 0; kk < 4; ++kk)
      acc = wmma_bf16(A[kk], b_frag(WcT, H_DIM, ct * 16, kk * 32, lane), acc);
    int col = ct * 16 + m;
    float bias = bc[col];
#pragma unroll
    for (int r = 0; r < 8; ++r)
      lt[w][(r + half * 8) * H_DIM + col] = (bf16)fmaxf(acc[r] + bias, 0.0f);
  }
  asm volatile("s_wait_dscnt 0x0" ::: "memory");

  // GEMM2: [16x128] x W1[128x64] -> relu -> lu
  v16bf A2[4];
#pragma unroll
  for (int kk = 0; kk < 4; ++kk) A2[kk] = a_frag_bf16(&lt[w][0], H_DIM, kk * 32, lane);
#pragma unroll
  for (int ct = 0; ct < 4; ++ct) {
    v8f acc = {};
#pragma unroll
    for (int kk = 0; kk < 4; ++kk)
      acc = wmma_bf16(A2[kk], b_frag(W1T, H_DIM, ct * 16, kk * 32, lane), acc);
    int col = ct * 16 + m;
    float bias = b1[col];
#pragma unroll
    for (int r = 0; r < 8; ++r)
      lu[w][(r + half * 8) * 64 + col] = (bf16)fmaxf(acc[r] + bias, 0.0f);
  }
  asm volatile("s_wait_dscnt 0x0" ::: "memory");

  // GEMM3: [16x64] x W2[64x32] -> relu -> lv
  v16bf A3[2];
#pragma unroll
  for (int kk = 0; kk < 2; ++kk) A3[kk] = a_frag_bf16(&lu[w][0], 64, kk * 32, lane);
#pragma unroll
  for (int ct = 0; ct < 2; ++ct) {
    v8f acc = {};
#pragma unroll
    for (int kk = 0; kk < 2; ++kk)
      acc = wmma_bf16(A3[kk], b_frag(W2T, 64, ct * 16, kk * 32, lane), acc);
    int col = ct * 16 + m;
    float bias = b2[col];
#pragma unroll
    for (int r = 0; r < 8; ++r)
      lv[w][(r + half * 8) * 32 + col] = (bf16)fmaxf(acc[r] + bias, 0.0f);
  }
  asm volatile("s_wait_dscnt 0x0" ::: "memory");

  // GEMM4: [16x32] x W3pad[32x32] -> sigmoid -> out[:, 0:18]
  v16bf A4 = a_frag_bf16(&lv[w][0], 32, 0, lane);
#pragma unroll
  for (int ct = 0; ct < 2; ++ct) {
    v8f acc = {};
    acc = wmma_bf16(A4, b_frag(W3T, 32, ct * 16, 0, lane), acc);
    int col = ct * 16 + m;
    float bias = (col < 18) ? b3[col] : 0.0f;
#pragma unroll
    for (int r = 0; r < 8; ++r) {
      if (col < 18) {
        float t = acc[r] + bias;
        out[(rowBase + r + half * 8) * 18 + col] = 1.0f / (1.0f + __expf(-t));
      }
    }
  }
}

// ---------------------------------------------------------------------------
// Launch
// ---------------------------------------------------------------------------
extern "C" void kernel_launch(void* const* d_in, const int* in_sizes, int n_in,
                              void* d_out, int out_size, void* d_ws, size_t ws_size,
                              hipStream_t stream) {
  const float* x     = (const float*)d_in[0];
  const float* We    = (const float*)d_in[1];
  const float* be    = (const float*)d_in[2];
  const float* gamma = (const float*)d_in[3];
  const float* beta  = (const float*)d_in[4];
  const float* mean  = (const float*)d_in[5];
  const float* var   = (const float*)d_in[6];
  const float* Wc    = (const float*)d_in[7];
  const float* bc    = (const float*)d_in[8];
  const float* W1    = (const float*)d_in[9];
  const float* b1    = (const float*)d_in[10];
  const float* W2    = (const float*)d_in[11];
  const float* b2    = (const float*)d_in[12];
  const float* W3    = (const float*)d_in[13];
  const float* b3    = (const float*)d_in[14];
  const int*   ei    = (const int*)d_in[15];
  float* out = (float*)d_out;

  // Workspace carve-up (256B aligned)
  char* ws = (char*)d_ws;
  size_t off = 0;
  auto carve = [&](size_t bytes) { size_t o = off; off = (off + bytes + 255) & ~(size_t)255; return o; };
  bf16*     h_bf = (bf16*)(ws + carve((size_t)N_NODES * H_DIM * sizeof(bf16)));
  float*    agg  = (float*)(ws + carve((size_t)N_NODES * H_DIM * sizeof(float)));
  unsigned* deg  = (unsigned*)(ws + carve((size_t)N_NODES * sizeof(unsigned)));
  float*    dinv = (float*)(ws + carve((size_t)N_NODES * sizeof(float)));
  bf16*     WeT  = (bf16*)(ws + carve(256 * 128 * sizeof(bf16)));
  bf16*     WcT  = (bf16*)(ws + carve(128 * 128 * sizeof(bf16)));
  bf16*     W1T  = (bf16*)(ws + carve(64 * 128 * sizeof(bf16)));
  bf16*     W2T  = (bf16*)(ws + carve(32 * 64 * sizeof(bf16)));
  bf16*     W3T  = (bf16*)(ws + carve(32 * 32 * sizeof(bf16)));

  const int tileBlocks = (N_TILES + 7) / 8;  // 8 waves (tiles) per 256-thread block

  prep_weights<<<(60416 + 255) / 256, 256, 0, stream>>>(We, Wc, W1, W2, W3,
                                                        WeT, WcT, W1T, W2T, W3T);
  init_deg<<<(N_NODES + 255) / 256, 256, 0, stream>>>(deg);
  embed_kernel<<<tileBlocks, 256, 0, stream>>>(x, WeT, be, gamma, beta, mean, var, h_bf);
  degree_kernel<<<(N_EDGES + 255) / 256, 256, 0, stream>>>(ei, deg);
  dinv_kernel<<<(N_NODES + 255) / 256, 256, 0, stream>>>(deg, dinv);
  agg_init<<<(N_NODES * H_DIM + 255) / 256, 256, 0, stream>>>(h_bf, dinv, agg);
  scatter_kernel<<<((size_t)N_EDGES * 32 + 255) / 256, 256, 0, stream>>>(ei, h_bf, dinv, agg);
  head_kernel<<<tileBlocks, 256, 0, stream>>>(agg, dinv, WcT, bc, W1T, b1,
                                              W2T, b2, W3T, b3, out);
}
